// GATModel_24232205484081
// MI455X (gfx1250) — matrix operand order repaired and checked by reading, not compiled, not run on hardware
//
#include <hip/hip_runtime.h>
#include <hip/hip_bf16.h>
#include <stdint.h>

#define LEAKY 0.2f

typedef __attribute__((ext_vector_type(16))) __bf16 v16bf;
typedef __attribute__((ext_vector_type(8)))  float  v8f;

// ---------------- utility ----------------

__device__ __forceinline__ void atomicMaxF(float* addr, float v) {
  int* ia = (int*)addr;
  int old = __float_as_int(*addr);
  while (__int_as_float(old) < v) {
    int assumed = old;
    old = atomicCAS(ia, assumed, __float_as_int(v));
    if (old == assumed) break;
  }
}

__global__ void k_fill(float* __restrict__ p, long n, float v) {
  long i = (long)blockIdx.x * blockDim.x + threadIdx.x;
  if (i < n) p[i] = v;
}

__global__ void k_fillbf(__bf16* __restrict__ p, long n) {
  long i = (long)blockIdx.x * blockDim.x + threadIdx.x;
  if (i < n) p[i] = (__bf16)0.f;
}

// ---------------- self loops: deg + mean incoming edge_attr ----------------

__global__ void k_deg(const int* __restrict__ ei, const float* __restrict__ ea,
                      float* __restrict__ deg, float* __restrict__ loop_sum, int E) {
  int e = blockIdx.x * blockDim.x + threadIdx.x;
  if (e >= E) return;
  int d = ei[E + e];
  atomicAdd(deg + d, 1.0f);
  atomicAdd(loop_sum + 2 * (long)d,     ea[2 * (long)e]);
  atomicAdd(loop_sum + 2 * (long)d + 1, ea[2 * (long)e + 1]);
}

__global__ void k_loopdiv(const float* __restrict__ deg, float* __restrict__ loopea, int N) {
  int n = blockIdx.x * blockDim.x + threadIdx.x;
  if (n >= N) return;
  float d = fmaxf(deg[n], 1.0f);
  loopea[2 * (long)n]     /= d;
  loopea[2 * (long)n + 1] /= d;
}

// ---------------- layer 1 linear: [N,2] @ [2,256] ----------------

__global__ void k_lin1(const float* __restrict__ x, const float* __restrict__ W1,
                       float* __restrict__ out, long total) {
  long t = (long)blockIdx.x * blockDim.x + threadIdx.x;
  if (t >= total) return;
  int f = (int)(t & 255);
  long n = t >> 8;
  out[t] = x[2 * n] * W1[f] + x[2 * n + 1] * W1[256 + f];
}

// ---------------- edge-attr attention coefficients s[k][h] ----------------

__global__ void k_se(const float* __restrict__ We, const float* __restrict__ ae,
                     float* __restrict__ se, int heads, int C) {
  int t = threadIdx.x;
  if (t >= 2 * heads) return;
  int k = t / heads, hd = t % heads;
  float s = 0.f;
  for (int c = 0; c < C; ++c) s += We[k * heads * C + hd * C + c] * ae[hd * C + c];
  se[t] = s;   // se[k*heads + hd]
}

// ---------------- per-node attention dots ----------------

__global__ void k_nodedots(const float* __restrict__ h, const float* __restrict__ a_s,
                           const float* __restrict__ a_d, float* __restrict__ asd,
                           float* __restrict__ addn, long total, int heads, int C) {
  long t = (long)blockIdx.x * blockDim.x + threadIdx.x;
  if (t >= total) return;
  int hd = (int)(t % heads);
  long n = t / heads;
  const float* hp = h + n * (long)(heads * C) + (long)hd * C;
  float s = 0.f, d = 0.f;
  for (int c = 0; c < C; ++c) {
    float v = hp[c];
    s += v * a_s[hd * C + c];
    d += v * a_d[hd * C + c];
  }
  asd[t] = s;
  addn[t] = d;
}

// ---------------- edge pass 1: alpha = leaky_relu(...), segment max ----------------

__global__ void k_edge1(const int* __restrict__ ei, const float* __restrict__ ea,
                        const float* __restrict__ loopea, const float* __restrict__ asd,
                        const float* __restrict__ addn, const float* __restrict__ se,
                        float* __restrict__ alpha, float* __restrict__ mx,
                        int E, int E2, int heads) {
  int e = blockIdx.x * blockDim.x + threadIdx.x;
  if (e >= E2) return;
  int s, d; float e0, e1;
  if (e < E) {
    s = ei[e]; d = ei[E + e];
    e0 = ea[2 * (long)e]; e1 = ea[2 * (long)e + 1];
  } else {
    s = d = e - E;
    e0 = loopea[2 * (long)s]; e1 = loopea[2 * (long)s + 1];
  }
  for (int hd = 0; hd < heads; ++hd) {
    float a = asd[(long)s * heads + hd] + addn[(long)d * heads + hd]
            + e0 * se[hd] + e1 * se[heads + hd];
    a = a > 0.f ? a : LEAKY * a;
    alpha[(long)e * heads + hd] = a;
    atomicMaxF(mx + (long)d * heads + hd, a);
  }
}

// ---------------- edge pass 2: exp + segment sum ----------------

__global__ void k_edge2(const int* __restrict__ ei, float* __restrict__ alpha,
                        const float* __restrict__ mx, float* __restrict__ den,
                        int E, int E2, int heads) {
  int e = blockIdx.x * blockDim.x + threadIdx.x;
  if (e >= E2) return;
  int d = (e < E) ? ei[E + e] : (e - E);
  for (int hd = 0; hd < heads; ++hd) {
    float ex = __expf(alpha[(long)e * heads + hd] - mx[(long)d * heads + hd]);
    alpha[(long)e * heads + hd] = ex;
    atomicAdd(den + (long)d * heads + hd, ex);
  }
}

// ---------------- edge pass 3: weighted scatter, one wave per edge ----------------
// 4 lanes compute per-head weights once; broadcast via wave shuffle.

__global__ void __launch_bounds__(256)
k_edge3(const int* __restrict__ ei, const float* __restrict__ alpha,
        const float* __restrict__ den, const float* __restrict__ hlin,
        float* __restrict__ out, int E, int E2, int heads, int C) {
  int e = blockIdx.x * 8 + (threadIdx.x >> 5);
  if (e >= E2) return;
  int lane = threadIdx.x & 31;
  int s, d;
  if (e < E) { s = ei[e]; d = ei[E + e]; } else { s = d = e - E; }
  float w = 0.f;
  if (lane < heads)
    w = alpha[(long)e * heads + lane] / den[(long)d * heads + lane];
  int F = heads * C;
  for (int f = lane; f < F; f += 32) {
    float wgt = __shfl(w, f / C, 32);
    atomicAdd(out + (long)d * F + f, wgt * hlin[(long)s * F + f]);
  }
}

// ---------------- bias + relu, fused f32 + bf16 dual write ----------------

__global__ void k_biasrelu(float* __restrict__ h, __bf16* __restrict__ hb,
                           const float* __restrict__ b, long total, int F) {
  long t = (long)blockIdx.x * blockDim.x + threadIdx.x;
  if (t >= total) return;
  float v = h[t] + b[(int)(t % F)];
  v = v > 0.f ? v : 0.f;
  h[t] = v;
  hb[t] = (__bf16)v;
}

// ---------------- pack weights [K,Nc] f32 -> WMMA B fragments bf16 ----------------
// packB[((nt*KT + kt)*32 + lane)*16 + i] = W[kt*32 + (lane>>4)*16 + i][nt*16 + (lane&15)]
// (B 32x16 in contiguous k order, matching k_gemm's A loads: lanes 0-15 hold K=0..15 of col N=lane,
//  lanes 16-31 hold K=16..31 of col N=lane-16)

__global__ void k_pack(const float* __restrict__ W, __bf16* __restrict__ P, int K, int Nc) {
  int t = blockIdx.x * blockDim.x + threadIdx.x;
  if (t >= K * Nc) return;
  int i    = t & 15;
  int lane = (t >> 4) & 31;
  int tile = t >> 9;
  int KT = K >> 5;
  int kt = tile % KT;
  int nt = tile / KT;
  int row = kt * 32 + (lane >> 4) * 16 + i;
  int col = nt * 16 + (lane & 15);
  P[t] = (__bf16)W[(long)row * Nc + col];
}

// ---------------- WMMA GEMM: C[M,Nc] = A[Mpad,K](bf16,row-major) x packed B --------
// block = 128 threads = 4 waves; block tile = 64 rows x 64 cols.
// Each wave: 64x16 strip = 4 accumulators, B fragment reused across 4 WMMAs.
// grid = (Mpad/64, Nc/64). A pad rows are zero; C stores guarded by M.

__global__ void __launch_bounds__(128)
k_gemm(const __bf16* __restrict__ A, const __bf16* __restrict__ Bp,
       float* __restrict__ C, int M, int K, int Nc) {
  __shared__ __bf16 sA[64 * 256];          // K <= 256 -> 32 KB
  const int tid = threadIdx.x;
  const long mt = blockIdx.x;

  // cooperative load: 64 full contiguous rows of A
  const int n16 = (64 * K) >> 3;           // number of uint4 (16B) chunks
  const uint4* gA = (const uint4*)(A + mt * 64 * (long)K);
  uint4* sA4 = (uint4*)sA;
  for (int i = tid; i < n16; i += 128) sA4[i] = gA[i];
  __syncthreads();

  const int wave = tid >> 5;
  const int lane = tid & 31;
  const int r  = lane & 15;                // row (A) / col (B,C) within tile
  const int hs = lane >> 4;                // half select
  const int nt = blockIdx.y * 4 + wave;    // 16-col tile index
  const int KT = K >> 5;

  v8f acc[4] = {};
  for (int kt = 0; kt < KT; ++kt) {
    // B fragment: contiguous 32B per lane, reused for 4 row tiles
    union { v16bf v; uint4 u[2]; } bfr;
    const uint4* bp = (const uint4*)(Bp + (((long)nt * KT + kt) * 32 + lane) * 16);
    bfr.u[0] = bp[0];
    bfr.u[1] = bp[1];
#pragma unroll
    for (int rt = 0; rt < 4; ++rt) {
      // A fragment (matches packed B): lane<16 -> K[kt*32+0..15];
      //                                lane>=16 -> K[kt*32+16..31]
      union { v16bf v; uint4 u[2]; } af;
      const uint4* ap = (const uint4*)(sA + (long)(rt * 16 + r) * K + kt * 32 + hs * 16);
      af.u[0] = ap[0];
      af.u[1] = ap[1];                     // +8 bf16 elements
      acc[rt] = __builtin_amdgcn_wmma_f32_16x16x32_bf16(
          false, af.v, false, bfr.v, (short)0, acc[rt], false, false);
    }
  }

  // C/D layout: VGPR i -> M = i + hs*8, N = lane&15
  const int col = nt * 16 + r;
#pragma unroll
  for (int rt = 0; rt < 4; ++rt) {
    long rowbase = mt * 64 + rt * 16 + hs * 8;
#pragma unroll
    for (int i = 0; i < 8; ++i) {
      long row = rowbase + i;
      if (row < M) C[row * Nc + col] = acc[rt][i];
    }
  }
}

// ---------------- final FC: [N,64] @ [64,3] + b ----------------

__global__ void k_final(const float* __restrict__ h, const float* __restrict__ Wf,
                        const float* __restrict__ bfv, float* __restrict__ out, int N) {
  int n = blockIdx.x * blockDim.x + threadIdx.x;
  if (n >= N) return;
  float a0 = bfv[0], a1 = bfv[1], a2 = bfv[2];
  const float* hp = h + (long)n * 64;
  for (int c = 0; c < 64; ++c) {
    float v = hp[c];
    a0 += v * Wf[c * 3];
    a1 += v * Wf[c * 3 + 1];
    a2 += v * Wf[c * 3 + 2];
  }
  out[(long)n * 3]     = a0;
  out[(long)n * 3 + 1] = a1;
  out[(long)n * 3 + 2] = a2;
}

// ---------------- host launcher ----------------

extern "C" void kernel_launch(void* const* d_in, const int* in_sizes, int n_in,
                              void* d_out, int out_size, void* d_ws, size_t ws_size,
                              hipStream_t stream) {
  const float* x   = (const float*)d_in[0];
  const int*   ei  = (const int*)d_in[1];
  const float* ea  = (const float*)d_in[2];
  const float* W1  = (const float*)d_in[3];
  const float* a1s = (const float*)d_in[4];
  const float* a1d = (const float*)d_in[5];
  const float* We1 = (const float*)d_in[6];
  const float* a1e = (const float*)d_in[7];
  const float* b1  = (const float*)d_in[8];
  const float* W2  = (const float*)d_in[9];
  const float* a2s = (const float*)d_in[10];
  const float* a2d = (const float*)d_in[11];
  const float* We2 = (const float*)d_in[12];
  const float* a2e = (const float*)d_in[13];
  const float* b2  = (const float*)d_in[14];
  const float* W3  = (const float*)d_in[15];
  const float* a3s = (const float*)d_in[16];
  const float* a3d = (const float*)d_in[17];
  const float* We3 = (const float*)d_in[18];
  const float* a3e = (const float*)d_in[19];
  const float* b3  = (const float*)d_in[20];
  const float* Wf  = (const float*)d_in[21];
  const float* bfv = (const float*)d_in[22];

  const int N    = in_sizes[0] / 2;   // x is [N,2]
  const int E    = in_sizes[2] / 2;   // edge_attr is [E,2]
  const int E2   = E + N;
  const int Mpad = (N + 63) & ~63;    // GEMM row padding (pad rows zero in bf16 buf)
  const int HD   = 4;                 // heads (layers 1-2)

  char* ws = (char*)d_ws;
  size_t off = 0;
  auto alloc = [&](size_t bytes) -> char* {
    char* p = ws + off;
    off += (bytes + 255) & ~(size_t)255;
    return p;
  };
  float*  bufA   = (float*) alloc((size_t)N * 256 * 4);
  float*  bufB   = (float*) alloc((size_t)N * 256 * 4);
  __bf16* hbf    = (__bf16*)alloc((size_t)Mpad * 256 * 2);
  __bf16* packB  = (__bf16*)alloc((size_t)256 * 256 * 2);
  float*  alpha  = (float*) alloc((size_t)E2 * HD * 4);
  float*  asd    = (float*) alloc((size_t)N * HD * 4);
  float*  addn   = (float*) alloc((size_t)N * HD * 4);
  float*  mx     = (float*) alloc((size_t)N * HD * 4);
  float*  den    = (float*) alloc((size_t)N * HD * 4);
  float*  deg    = (float*) alloc((size_t)N * 4);
  float*  loopea = (float*) alloc((size_t)N * 2 * 4);
  float*  se     = (float*) alloc(256);

  auto nb = [](long n) { return dim3((unsigned)((n + 255) / 256)); };

  // zero the bf16 pad rows (deterministic, every call)
  if (Mpad > N)
    k_fillbf<<<nb((long)(Mpad - N) * 256), 256, 0, stream>>>(
        hbf + (long)N * 256, (long)(Mpad - N) * 256);

  // self loops
  k_fill<<<nb(N), 256, 0, stream>>>(deg, N, 0.f);
  k_fill<<<nb(2L * N), 256, 0, stream>>>(loopea, 2L * N, 0.f);
  k_deg<<<nb(E), 256, 0, stream>>>(ei, ea, deg, loopea, E);
  k_loopdiv<<<nb(N), 256, 0, stream>>>(deg, loopea, N);

  // layer 1 linear
  k_lin1<<<nb((long)N * 256), 256, 0, stream>>>(x, W1, bufA, (long)N * 256);

  auto run_attn = [&](const float* hlin, float* outp, const float* asp, const float* adp,
                      const float* Wep, const float* aep, const float* bp, int heads, int C) {
    long F = (long)heads * C;
    k_se<<<1, 32, 0, stream>>>(Wep, aep, se, heads, C);
    k_nodedots<<<nb((long)N * heads), 256, 0, stream>>>(hlin, asp, adp, asd, addn,
                                                        (long)N * heads, heads, C);
    k_fill<<<nb((long)N * heads), 256, 0, stream>>>(mx, (long)N * heads, -3.0e38f);
    k_fill<<<nb((long)N * heads), 256, 0, stream>>>(den, (long)N * heads, 0.f);
    k_fill<<<nb((long)N * F), 256, 0, stream>>>(outp, (long)N * F, 0.f);
    k_edge1<<<nb(E2), 256, 0, stream>>>(ei, ea, loopea, asd, addn, se, alpha, mx, E, E2, heads);
    k_edge2<<<nb(E2), 256, 0, stream>>>(ei, alpha, mx, den, E, E2, heads);
    k_edge3<<<dim3((unsigned)((E2 + 7) / 8)), 256, 0, stream>>>(ei, alpha, den, hlin, outp,
                                                                E, E2, heads, C);
    // fused bias + relu + bf16 convert (bf16 copy feeds the next WMMA GEMM)
    k_biasrelu<<<nb((long)N * F), 256, 0, stream>>>(outp, hbf, bp, (long)N * F, (int)F);
  };

  auto run_gemm = [&](const float* W, float* Cout, int K, int Nc) {
    k_pack<<<nb((long)K * Nc), 256, 0, stream>>>(W, packB, K, Nc);
    k_gemm<<<dim3((unsigned)(Mpad / 64), (unsigned)(Nc / 64)), 128, 0, stream>>>(
        hbf, packB, Cout, N, K, Nc);
  };

  // layer 1 attention (heads=4, C=64, concat)
  run_attn(bufA, bufB, a1s, a1d, We1, a1e, b1, 4, 64);

  // layer 2: WMMA GEMM [N,256]@[256,256] (grid.y = 256/64 = 4 covers all columns)
  run_gemm(W2, bufA, 256, 256);
  run_attn(bufA, bufB, a2s, a2d, We2, a2e, b2, 4, 64);

  // layer 3: WMMA GEMM [N,256]@[256,64]; heads=1, concat=False (mean over 1 head = id)
  run_gemm(W3, bufA, 256, 64);
  run_attn(bufA, bufB, a3s, a3d, We3, a3e, b3, 1, 64);

  // final FC
  k_final<<<nb(N), 256, 0, stream>>>(bufB, Wf, bfv, (float*)d_out, N);
}
